// RecursiveDecoder_31233002176707
// MI455X (gfx1250) — compile-verified, matrix operand order checked by reading, not run on hardware
//
#include <hip/hip_runtime.h>
#include <hip/hip_bf16.h>

// ---------------------------------------------------------------------------
// CDNA5 WMMA (wave32): v_wmma_f32_16x16x32_bf16
// ---------------------------------------------------------------------------
typedef __bf16 bf16_t;
typedef bf16_t v16bf __attribute__((ext_vector_type(16)));
typedef float  v8f   __attribute__((ext_vector_type(8)));
typedef unsigned int v4u __attribute__((ext_vector_type(4)));

union Frag { v16bf v; v4u q[2]; unsigned short u[16]; };

__device__ __forceinline__ unsigned short f2bf(float f) {
    union { float f; unsigned int i; } c; c.f = f;
    unsigned int u = c.i;
    u += 0x7FFFu + ((u >> 16) & 1u);   // round-to-nearest-even to bf16
    return (unsigned short)(u >> 16);
}

// ---------------------------------------------------------------------------
// WMMA GEMM, bf16 operands: C = act(A @ B + bias)  [+ optional bf16 copy]
//   Abf : bf16 row-major (M x K), lda = K stride
//   Btb : bf16 TRANSPOSED (Npad x K), ldb = K stride, zero-padded cols
// Fragment layout (ISA 7.12.2): lane L holds row/col L%16,
// k = 8*(L/16) + {0..7} and {16..23}  -> two contiguous b128 loads.
// ---------------------------------------------------------------------------
__global__ __launch_bounds__(32) void gemm_wmma_kernel(
    const unsigned short* __restrict__ Abf, int lda,
    const unsigned short* __restrict__ Btb, int ldb,
    const float* __restrict__ bias,
    float* __restrict__ C, int ldc,
    unsigned short* __restrict__ Cbf, int ldcb,
    int N, int Kc, int do_relu)
{
    const int lane = threadIdx.x & 31;
    const int m0 = blockIdx.y << 4;
    const int n0 = blockIdx.x << 4;
    const int l16 = lane & 15;
    const int koff = (lane >> 4) << 3;

    const unsigned short* arow = Abf + (size_t)(m0 + l16) * lda + koff;
    const unsigned short* brow = Btb + (size_t)(n0 + l16) * ldb + koff;

    v8f acc = {};
    for (int k0 = 0; k0 < Kc; k0 += 32) {
        Frag af, bf;
        af.q[0] = *(const v4u*)(arow + k0);
        af.q[1] = *(const v4u*)(arow + k0 + 16);
        bf.q[0] = *(const v4u*)(brow + k0);
        bf.q[1] = *(const v4u*)(brow + k0 + 16);
        acc = __builtin_amdgcn_wmma_f32_16x16x32_bf16(
            false, af.v, false, bf.v, (short)0, acc, false, false);
    }

    const int colC = n0 + l16;
    if (colC < N) {
        const int rbase = m0 + ((lane >> 4) << 3);
        const float bb = bias ? bias[colC] : 0.0f;
#pragma unroll
        for (int v = 0; v < 8; ++v) {
            float x = acc[v] + bb;
            if (do_relu) x = x > 0.0f ? x : 0.0f;
            C[(rbase + v) * ldc + colC] = x;
            if (Cbf) Cbf[(rbase + v) * ldcb + colC] = f2bf(x);
        }
    }
}

// ---------------------------------------------------------------------------
// Weight preconversion: fp32 (K x N) -> bf16 transposed (Npad x K), zero pad.
// ---------------------------------------------------------------------------
__global__ __launch_bounds__(256) void cvt_bt_kernel(
    const float* __restrict__ B, int N, int Kc,
    unsigned short* __restrict__ out)
{
    const int idx = blockIdx.x * 256 + threadIdx.x;   // over Npad*Kc
    const int n = idx / Kc;
    const int k = idx - n * Kc;
    out[idx] = (n < N) ? f2bf(B[k * N + n]) : (unsigned short)0;
}

// ---------------------------------------------------------------------------
// cf = relu(pf @ Wp + bp); writes fp32 + bf16 copies.
// ---------------------------------------------------------------------------
__global__ __launch_bounds__(256) void cf_kernel(
    const float* __restrict__ pf, const float* __restrict__ Wp,
    const float* __restrict__ bp, float* __restrict__ cf,
    unsigned short* __restrict__ cf_bf)
{
    __shared__ float spf[256];
    const int col = blockIdx.x * 256 + threadIdx.x;
    spf[threadIdx.x] = pf[threadIdx.x];
    __syncthreads();
    float s = bp[col];
    for (int f = 0; f < 256; ++f) s += spf[f] * Wp[f * 65536 + col];
    s = s > 0.0f ? s : 0.0f;
    cf[col] = s;
    cf_bf[col] = f2bf(s);
}

__global__ __launch_bounds__(256) void exists_kernel(
    const float* __restrict__ cf, const float* __restrict__ We,
    const float* __restrict__ be, float* __restrict__ ex)
{
    __shared__ float sWe[256];
    const int i = threadIdx.x;
    sWe[i] = We[i];
    __syncthreads();
    float s = be[0];
    for (int h = 0; h < 256; ++h) s += cf[i * 256 + h] * sWe[h];
    ex[i] = s;
}

// ---------------------------------------------------------------------------
// eel[i,j,t] = relu(U[i]+V[j]+bel) . Wee[t] + bee[t]
// ---------------------------------------------------------------------------
__global__ __launch_bounds__(256) void eel_kernel(
    const float* __restrict__ U, const float* __restrict__ V,
    const float* __restrict__ bel, const float* __restrict__ Wee,
    const float* __restrict__ bee, float* __restrict__ eel)
{
    __shared__ float sUi[256], sBel[256], sW0[256], sW1[256];
    const int i = blockIdx.x;
    const int j = threadIdx.x;
    sUi[j] = U[i * 256 + j];
    sBel[j] = bel[j];
    sW0[j] = Wee[j];
    sW1[j] = Wee[256 + j];
    __syncthreads();
    float s0 = bee[0], s1 = bee[1];
    for (int d = 0; d < 256; ++d) {
        float l = sUi[d] + V[j * 256 + d] + sBel[d];
        l = l > 0.0f ? l : 0.0f;
        s0 += l * sW0[d];
        s1 += l * sW1[d];
    }
    eel[(i * 256 + j) * 2 + 0] = s0;
    eel[(i * 256 + j) * 2 + 1] = s1;
}

// ---------------------------------------------------------------------------
// Fused message-pass iteration. Block = row i, 256 threads (8 waves).
// Per 16-wide j tile:
//  (A) async-copy the contiguous 16KB block V[j0*256 ..) into LDS
//      (global_load_async_to_lds_b128, ASYNCcnt), build lat tile bf16,
//  (B) WMMA lat@Wl via ds_load_b128 / b128 of transposed bf16 Wl,
//  (C) masked max over (j,t).
// ---------------------------------------------------------------------------
__global__ __launch_bounds__(256) void msgmax_kernel(
    const float* __restrict__ U, const float* __restrict__ V,
    const float* __restrict__ bel,
    const unsigned short* __restrict__ WlT,   // bf16 transposed (256 x 256)
    const float* __restrict__ Wt,             // (2,256)
    const float* __restrict__ bop,
    const float* __restrict__ avec, const float* __restrict__ bvec,
    const float* __restrict__ eel,  const float* __restrict__ ex,
    float* __restrict__ outf, unsigned short* __restrict__ outf_bf)
{
    __shared__ float sUi[256];
    __shared__ float sBel[256];
    __shared__ __align__(16) float sVtile[16][256];        // async-staged V, 16 KB
    __shared__ __align__(16) unsigned short sLat[16][256]; // bf16 lat tile, 8 KB
    __shared__ float sClat[16][256];                       // c_lat tile, 16 KB

    const int i = blockIdx.x;
    const int h = threadIdx.x;

    sUi[h]  = U[i * 256 + h];
    sBel[h] = bel[h];
    const float a_h   = avec[i * 256 + h];
    const float bop_h = bop[h];
    const float wt0   = Wt[h];
    const float wt1   = Wt[256 + h];
    const bool  ok_i  = ex[i] > 0.0f;
    float acc = 0.0f;

    // LDS byte offset of sVtile: generic shared address keeps the LDS offset
    // in its low 32 bits (ISA 10.2).
    const unsigned lds_v = (unsigned)(unsigned long long)(void*)&sVtile[0][0];
    __syncthreads();

    for (int jb = 0; jb < 16; ++jb) {
        const int j0 = jb * 16;

        // Phase A0: async stage 16 contiguous V rows (16 KB) into LDS.
        {
            const unsigned long long gbase =
                (unsigned long long)(const void*)(V + (size_t)j0 * 256);
#pragma unroll
            for (int r = 0; r < 4; ++r) {
                const unsigned la = lds_v + (unsigned)(r * 4096 + h * 16);
                const unsigned long long ga =
                    gbase + (unsigned long long)(r * 4096 + h * 16);
                asm volatile("global_load_async_to_lds_b128 %0, %1, off"
                             :: "v"(la), "v"(ga) : "memory");
            }
            asm volatile("s_wait_asynccnt 0x0" ::: "memory");
        }
        __syncthreads();

        // Phase A1: lat tile (16 j x 256 d) -> bf16 LDS
        {
            const int jl = h >> 4;
            const int d0 = (h & 15) << 4;
#pragma unroll
            for (int e = 0; e < 16; ++e) {
                const int d = d0 + e;
                float l = sUi[d] + sVtile[jl][d] + sBel[d];
                l = l > 0.0f ? l : 0.0f;
                sLat[jl][d] = f2bf(l);
            }
        }
        __syncthreads();

        // Phase B: c_lat tile = lat(16x256) @ Wl(256x256); 8 waves x 2 n-tiles
        {
            const int lane = h & 31;
            const int l16 = lane & 15;
            const int koff = (lane >> 4) << 3;
            const unsigned short* arow = &sLat[l16][koff];
#pragma unroll
            for (int nt = 0; nt < 2; ++nt) {
                const int n0 = (h >> 5) * 32 + nt * 16;
                const unsigned short* brow = WlT + (size_t)(n0 + l16) * 256 + koff;
                v8f c8 = {};
#pragma unroll
                for (int kk = 0; kk < 8; ++kk) {
                    Frag af, bf;
                    af.q[0] = *(const v4u*)(arow + kk * 32);
                    af.q[1] = *(const v4u*)(arow + kk * 32 + 16);
                    bf.q[0] = *(const v4u*)(brow + kk * 32);
                    bf.q[1] = *(const v4u*)(brow + kk * 32 + 16);
                    c8 = __builtin_amdgcn_wmma_f32_16x16x32_bf16(
                        false, af.v, false, bf.v, (short)0, c8, false, false);
                }
                const int rbase = (lane >> 4) << 3;
#pragma unroll
                for (int v = 0; v < 8; ++v)
                    sClat[rbase + v][n0 + l16] = c8[v];
            }
        }
        __syncthreads();

        // Phase C: masked max over (j in tile, t)
        if (ok_i) {
            for (int jl = 0; jl < 16; ++jl) {
                const int j = j0 + jl;
                if (ex[j] > 0.0f) {
                    const float base = a_h + bvec[j * 256 + h] + sClat[jl][h] + bop_h;
                    const float e0 = eel[(i * 256 + j) * 2 + 0];
                    const float e1 = eel[(i * 256 + j) * 2 + 1];
                    if (e0 > 0.0f) {
                        float m = base + e0 * wt0;
                        m = m > 0.0f ? m : 0.0f;
                        acc = acc > m ? acc : m;
                    }
                    if (e1 > 0.0f) {
                        float m = base + e1 * wt1;
                        m = m > 0.0f ? m : 0.0f;
                        acc = acc > m ? acc : m;
                    }
                }
            }
        }
        __syncthreads();
    }
    outf[i * 256 + h] = acc;
    if (outf_bf) outf_bf[i * 256 + h] = f2bf(acc);
}

// allf_bf = [cf | f1 | f2] as bf16  (K x 768)
__global__ __launch_bounds__(256) void pack_kernel(
    const float* __restrict__ cf, const float* __restrict__ f1,
    const float* __restrict__ f2, unsigned short* __restrict__ allf_bf)
{
    const int i = blockIdx.x, t = threadIdx.x;
    allf_bf[i * 768 + t]       = f2bf(cf[i * 256 + t]);
    allf_bf[i * 768 + 256 + t] = f2bf(f1[i * 256 + t]);
    allf_bf[i * 768 + 512 + t] = f2bf(f2[i * 256 + t]);
}

// ---------------------------------------------------------------------------
extern "C" void kernel_launch(void* const* d_in, const int* in_sizes, int n_in,
                              void* d_out, int out_size, void* d_ws, size_t ws_size,
                              hipStream_t stream) {
    const float* pf  = (const float*)d_in[0];
    const float* Wp  = (const float*)d_in[1];
    const float* bp  = (const float*)d_in[2];
    const float* We  = (const float*)d_in[3];
    const float* be  = (const float*)d_in[4];
    const float* Wel = (const float*)d_in[5];   // (512,256)
    const float* bel = (const float*)d_in[6];
    const float* Wee = (const float*)d_in[7];   // (2,256)
    const float* bee = (const float*)d_in[8];
    const float* Wop = (const float*)d_in[9];   // (770,256)
    const float* bop = (const float*)d_in[10];
    const float* Wc  = (const float*)d_in[11];  // (768,256)
    const float* bc  = (const float*)d_in[12];
    const float* Ws  = (const float*)d_in[13];  // (256,57)
    const float* bs  = (const float*)d_in[14];
    const float* Wc2 = (const float*)d_in[15];  // (256,256)
    const float* bc2 = (const float*)d_in[16];

    const float* Wa = Wop;                 // rows 0..255
    const float* Wb = Wop + 256 * 256;     // rows 256..511
    const float* Wl = Wop + 512 * 256;     // rows 512..767
    const float* Wt = Wop + 768 * 256;     // rows 768..769

    // fp32 scratch
    float* fws = (float*)d_ws;
    float* cf  = fws;                 // 65536
    float* U   = cf + 65536;
    float* V   = U + 65536;
    float* av  = V + 65536;
    float* bv  = av + 65536;
    float* f1  = bv + 65536;
    float* f2  = f1 + 65536;
    float* hid = f2 + 65536;          // 65536 -> total 524288 floats (2 MB)

    // bf16 scratch (16B-aligned: 2 MB offset)
    unsigned short* bws   = (unsigned short*)(fws + 524288);
    unsigned short* WelTt = bws;                  // 65536 (top, transposed)
    unsigned short* WelTb = WelTt + 65536;        // 65536
    unsigned short* WaT   = WelTb + 65536;        // 65536
    unsigned short* WbT   = WaT + 65536;          // 65536
    unsigned short* WlT   = WbT + 65536;          // 65536
    unsigned short* Wc2T  = WlT + 65536;          // 65536
    unsigned short* WcT   = Wc2T + 65536;         // 256*768 = 196608
    unsigned short* WsT   = WcT + 196608;         // 64*256 = 16384
    unsigned short* cf_bf = WsT + 16384;          // 65536
    unsigned short* f1_bf = cf_bf + 65536;        // 65536
    unsigned short* af_bf = f1_bf + 65536;        // 196608 (allf)
    unsigned short* hd_bf = af_bf + 196608;       // 65536

    float* out_child  = (float*)d_out;            // 256*256
    float* out_sem    = out_child + 65536;        // 256*57
    float* out_exists = out_sem + 256 * 57;       // 256
    float* out_eel    = out_exists + 256;         // 256*256*2

    const dim3 g16(16, 16);

    // Weight preconversion to bf16 transposed (zero-padded)
    cvt_bt_kernel<<<256, 256, 0, stream>>>(Wel,             256, 256, WelTt);
    cvt_bt_kernel<<<256, 256, 0, stream>>>(Wel + 256 * 256, 256, 256, WelTb);
    cvt_bt_kernel<<<256, 256, 0, stream>>>(Wa,  256, 256, WaT);
    cvt_bt_kernel<<<256, 256, 0, stream>>>(Wb,  256, 256, WbT);
    cvt_bt_kernel<<<256, 256, 0, stream>>>(Wl,  256, 256, WlT);
    cvt_bt_kernel<<<256, 256, 0, stream>>>(Wc2, 256, 256, Wc2T);
    cvt_bt_kernel<<<768, 256, 0, stream>>>(Wc,  256, 768, WcT);   // Npad=256,K=768
    cvt_bt_kernel<<<64,  256, 0, stream>>>(Ws,   57, 256, WsT);   // Npad=64

    // cf, exists
    cf_kernel<<<256, 256, 0, stream>>>(pf, Wp, bp, cf, cf_bf);
    exists_kernel<<<1, 256, 0, stream>>>(cf, We, be, out_exists);

    // U = cf @ Wel_top, V = cf @ Wel_bot
    gemm_wmma_kernel<<<g16, 32, 0, stream>>>(cf_bf, 256, WelTt, 256, nullptr, U, 256, nullptr, 0, 256, 256, 0);
    gemm_wmma_kernel<<<g16, 32, 0, stream>>>(cf_bf, 256, WelTb, 256, nullptr, V, 256, nullptr, 0, 256, 256, 0);

    // eel output
    eel_kernel<<<256, 256, 0, stream>>>(U, V, bel, Wee, bee, out_eel);

    // iteration 1
    gemm_wmma_kernel<<<g16, 32, 0, stream>>>(cf_bf, 256, WaT, 256, nullptr, av, 256, nullptr, 0, 256, 256, 0);
    gemm_wmma_kernel<<<g16, 32, 0, stream>>>(cf_bf, 256, WbT, 256, nullptr, bv, 256, nullptr, 0, 256, 256, 0);
    msgmax_kernel<<<256, 256, 0, stream>>>(U, V, bel, WlT, Wt, bop, av, bv, out_eel, out_exists, f1, f1_bf);

    // iteration 2
    gemm_wmma_kernel<<<g16, 32, 0, stream>>>(f1_bf, 256, WaT, 256, nullptr, av, 256, nullptr, 0, 256, 256, 0);
    gemm_wmma_kernel<<<g16, 32, 0, stream>>>(f1_bf, 256, WbT, 256, nullptr, bv, 256, nullptr, 0, 256, 256, 0);
    msgmax_kernel<<<256, 256, 0, stream>>>(U, V, bel, WlT, Wt, bop, av, bv, out_eel, out_exists, f2, nullptr);

    // heads
    pack_kernel<<<256, 256, 0, stream>>>(cf, f1, f2, af_bf);
    gemm_wmma_kernel<<<g16, 32, 0, stream>>>(af_bf, 768, WcT, 768, bc, hid, 256, hd_bf, 256, 256, 768, 1);
    gemm_wmma_kernel<<<dim3(4, 16), 32, 0, stream>>>(hd_bf, 256, WsT, 256, bs, out_sem, 57, nullptr, 0, 57, 256, 0);
    gemm_wmma_kernel<<<g16, 32, 0, stream>>>(hd_bf, 256, Wc2T, 256, bc2, out_child, 256, nullptr, 0, 256, 256, 1);
}
